// MLA_69681549410912
// MI455X (gfx1250) — matrix operand order repaired and checked by reading, not compiled, not run on hardware
//
#include <hip/hip_runtime.h>
#include <stdint.h>

typedef __attribute__((ext_vector_type(16))) _Float16 v16h;
typedef __attribute__((ext_vector_type(8)))  float    v8f;

#define B_    2
#define S_    2048
#define H_    2048
#define NH_   16
#define DH_   128
#define DR_   64
#define DHV_  128
#define DCQ_  768
#define DCKV_ 512
#define DQK_  192          // DH + DR
#define SCALE_F 0.07216878364870323f   // (DR+DH)^-0.5 = 192^-0.5

__device__ __forceinline__ v8f wmma16(v16h a, v16h b, v8f c) {
  // D = A(16x32 f16) * B(32x16 f16) + C(16x16 f32)
  return __builtin_amdgcn_wmma_f32_16x16x32_f16(false, a, false, b, (short)0, c, false, false);
}

union AF { v16h v; uint32_t u[8]; };

// Async copy 16B per lane: global (saddr + 32-bit voffset) -> LDS (32-bit offset).
__device__ __forceinline__ void async_b128(uint32_t lds_off, const void* sbase, uint32_t goff) {
  asm volatile("global_load_async_to_lds_b128 %0, %1, %2"
               : : "v"(lds_off), "v"(goff), "s"(sbase) : "memory");
}
template<int N>
__device__ __forceinline__ void s_wait_async() {
  asm volatile("s_wait_asynccnt %0" : : "n"(N) : "memory");
}
__device__ __forceinline__ uint32_t lds_off32(const void* p) {
  return (uint32_t)(uintptr_t)p;   // low 32 bits of generic shared ptr = LDS offset
}

// ---------------------------------------------------------------- converts
__global__ __launch_bounds__(256)
void cvt_f32_f16(const float* __restrict__ src, _Float16* __restrict__ dst, size_t n) {
  size_t i = (size_t)blockIdx.x * 256 + threadIdx.x;
  if (i < n) dst[i] = (_Float16)src[i];
}

// W[K,N] f32 -> W^T[N,K] f16 (so GEMM B-tiles are plain row copies)
__global__ __launch_bounds__(256)
void cvt_f32_f16_t(const float* __restrict__ src, _Float16* __restrict__ dst, int K, int N) {
  size_t i = (size_t)blockIdx.x * 256 + threadIdx.x;
  if (i >= (size_t)K * N) return;
  int n = (int)(i % N);
  int k = (int)(i / N);
  dst[(size_t)n * K + k] = (_Float16)src[i];
}

// ---------------------------------------------------------------- GEMM
// C[M,N] = A[M,K] @ W[K,N] + bias[N], W given transposed (Bt[N,K], f16 row-major).
// 64x64 tile, 256 threads = 8 waves (4 M-quadrants x 2 N-halves, 16x32 each).
// Double-buffered async global->LDS staging (2 async instr / tile / wave).
template<bool OUT_F32>
__global__ __launch_bounds__(256)
void gemm64(const _Float16* __restrict__ A, const _Float16* __restrict__ Bt,
            const float* __restrict__ bias, void* __restrict__ Cv,
            int M, int N, int K)
{
  __shared__ _Float16 sA[2][64 * 32];   // [m][k]
  __shared__ _Float16 sB[2][64 * 32];   // [n][k]
  const int tid   = threadIdx.x;
  const int lane  = tid & 31;
  const int wid   = tid >> 5;
  const int waveM = wid >> 1;
  const int waveN = wid & 1;
  const int row16 = lane & 15;
  const int hi    = lane >> 4;
  const int bm    = blockIdx.y * 64;
  const int bn    = blockIdx.x * 64;

  const int crow = tid >> 2;        // 0..63 : tile row (m or n)
  const int ccu  = tid & 3;         // 0..3  : 16B chunk within 64B row
  const uint32_t aOff[2] = { lds_off32(&sA[0][0]), lds_off32(&sA[1][0]) };
  const uint32_t bOff[2] = { lds_off32(&sB[0][0]), lds_off32(&sB[1][0]) };
  const uint32_t lchunk  = (uint32_t)(crow * 64 + ccu * 16);

  auto issue = [&](int buf, int k0) {
    async_b128(aOff[buf] + lchunk, A,  (uint32_t)(((size_t)(bm + crow) * K + k0) * 2) + ccu * 16);
    async_b128(bOff[buf] + lchunk, Bt, (uint32_t)(((size_t)(bn + crow) * K + k0) * 2) + ccu * 16);
  };

  v8f acc[2] = {};
  issue(0, 0);
  int cur = 0;
  for (int k0 = 0; k0 < K; k0 += 32, cur ^= 1) {
    if (k0 + 32 < K) { issue(cur ^ 1, k0 + 32); s_wait_async<2>(); }
    else             { s_wait_async<0>(); }
    __syncthreads();
    // A fragment: row = lane&15, K = (i&7) + 8*hi + 16*(i>=8)
    AF a;
    {
      const uint32_t* pA = reinterpret_cast<const uint32_t*>(&sA[cur][0] + (waveM * 16 + row16) * 32);
      #pragma unroll
      for (int p = 0; p < 8; ++p) a.u[p] = pA[p + (p >= 4 ? 4 : 0) + hi * 4];
    }
    #pragma unroll
    for (int t = 0; t < 2; ++t) {
      // B fragment: N = col, K = (i&7) + 8*hi + 16*(i>=8)
      AF b;
      const int col = waveN * 32 + t * 16 + row16;
      const uint32_t* pB = reinterpret_cast<const uint32_t*>(&sB[cur][0] + col * 32);
      #pragma unroll
      for (int p = 0; p < 8; ++p) b.u[p] = pB[p + (p >= 4 ? 4 : 0) + hi * 4];
      acc[t] = wmma16(a.v, b.v, acc[t]);
    }
    __syncthreads();
  }
  #pragma unroll
  for (int t = 0; t < 2; ++t) {
    #pragma unroll
    for (int r = 0; r < 8; ++r) {
      const int rg = bm + waveM * 16 + r + hi * 8;   // D: M = r + 8*hi
      const int cg = bn + waveN * 32 + t * 16 + row16;
      const float v = acc[t][r] + bias[cg];
      if (OUT_F32) reinterpret_cast<float*>(Cv)[(size_t)rg * N + cg] = v;
      else         reinterpret_cast<_Float16*>(Cv)[(size_t)rg * N + cg] = (_Float16)v;
    }
  }
}

// ---------------------------------------------------------------- q/k/v assembly (+RoPE)
__global__ __launch_bounds__(256)
void build_q(const _Float16* __restrict__ qc, const _Float16* __restrict__ qr,
             const float* __restrict__ freq, _Float16* __restrict__ q)
{
  const int total = B_ * NH_ * S_ * DQK_;
  int idx = blockIdx.x * 256 + threadIdx.x;
  if (idx >= total) return;
  int d = idx % DQK_; int t = idx / DQK_;
  int s = t % S_; t /= S_;
  int h = t % NH_; int b = t / NH_;
  const size_t row = (size_t)b * S_ + s;
  _Float16 val;
  if (d < DH_) {
    val = qc[row * (NH_ * DH_) + h * DH_ + d];
  } else {
    int dr = d - DH_;
    int i  = dr >> 1;
    float f  = freq[s * (DR_ / 2) + i];
    float c  = __cosf(f), sn = __sinf(f);
    float x1 = (float)qr[row * (NH_ * DR_) + h * DR_ + 2 * i];
    float x2 = (float)qr[row * (NH_ * DR_) + h * DR_ + 2 * i + 1];
    float o  = (dr & 1) ? (x1 * sn + x2 * c) : (x1 * c - x2 * sn);
    val = (_Float16)o;
  }
  q[idx] = val;
}

__global__ __launch_bounds__(256)
void build_k(const _Float16* __restrict__ kvc, const _Float16* __restrict__ krr,
             const float* __restrict__ freq, _Float16* __restrict__ k)
{
  const int total = B_ * NH_ * S_ * DQK_;
  int idx = blockIdx.x * 256 + threadIdx.x;
  if (idx >= total) return;
  int d = idx % DQK_; int t = idx / DQK_;
  int s = t % S_; t /= S_;
  int h = t % NH_; int b = t / NH_;
  const size_t row = (size_t)b * S_ + s;
  _Float16 val;
  if (d < DH_) {
    val = kvc[row * (NH_ * (DH_ + DHV_)) + h * (DH_ + DHV_) + d];
  } else {
    int dr = d - DH_;
    int i  = dr >> 1;
    float f  = freq[s * (DR_ / 2) + i];
    float c  = __cosf(f), sn = __sinf(f);
    float x1 = (float)krr[row * DR_ + 2 * i];
    float x2 = (float)krr[row * DR_ + 2 * i + 1];
    float o  = (dr & 1) ? (x1 * sn + x2 * c) : (x1 * c - x2 * sn);
    val = (_Float16)o;
  }
  k[idx] = val;
}

// V transposed to [b, h, n, s] so flash V-tiles are plain strided row copies.
__global__ __launch_bounds__(256)
void build_vt(const _Float16* __restrict__ kvc, _Float16* __restrict__ vt)
{
  const int total = B_ * NH_ * DHV_ * S_;
  int idx = blockIdx.x * 256 + threadIdx.x;
  if (idx >= total) return;
  int s = idx % S_; int t = idx / S_;
  int n = t % DHV_; t /= DHV_;
  int h = t % NH_; int b = t / NH_;
  const size_t row = (size_t)b * S_ + s;
  vt[idx] = kvc[row * (NH_ * (DH_ + DHV_)) + h * (DH_ + DHV_) + DH_ + n];
}

// ---------------------------------------------------------------- flash attention
// Block = 128 threads (4 waves), 64 query rows of one (b,h); wave owns 16 rows.
// Key blocks of 64, causal. Double-buffered async K/V staging
// (20 async instr / tile / wave: 12 K + 8 V).
__global__ __launch_bounds__(128)
void flash_attn(const _Float16* __restrict__ Q, const _Float16* __restrict__ Kt,
                const _Float16* __restrict__ Vt, _Float16* __restrict__ O)
{
  __shared__ _Float16 sK[2][64 * DQK_];   // [key][192]
  __shared__ _Float16 sV[2][DHV_ * 64];   // [n][key]
  __shared__ _Float16 sP[4 * 16 * 64];    // per-wave P staging

  const int tid   = threadIdx.x;
  const int lane  = tid & 31;
  const int wid   = tid >> 5;
  const int row16 = lane & 15;
  const int hi    = lane >> 4;
  const int q0    = blockIdx.x * 64;
  const int h     = blockIdx.y;
  const int b     = blockIdx.z;

  const size_t qkbase = ((size_t)(b * NH_ + h)) * S_ * DQK_;
  const size_t vtbase = ((size_t)(b * NH_ + h)) * DHV_ * S_;
  const uint32_t kOff[2] = { lds_off32(&sK[0][0]), lds_off32(&sK[1][0]) };
  const uint32_t vOff[2] = { lds_off32(&sV[0][0]), lds_off32(&sV[1][0]) };

  auto issueK = [&](int buf, int s0) {          // 64x192 contiguous rows: 24KB
    const uint32_t g0 = (uint32_t)((qkbase + (size_t)s0 * DQK_) * 2);
    #pragma unroll
    for (int c = 0; c < 12; ++c) {
      const uint32_t o = (uint32_t)(tid + c * 128) * 16;
      async_b128(kOff[buf] + o, Kt, g0 + o);
    }
  };
  auto issueV = [&](int buf, int s0) {          // 128 rows x 64 keys, stride S
    #pragma unroll
    for (int c = 0; c < 8; ++c) {
      const int j = tid + c * 128;
      const int n = j >> 3, cu = j & 7;
      const uint32_t go = (uint32_t)((vtbase + (size_t)n * S_ + s0) * 2) + cu * 16;
      async_b128(vOff[buf] + (uint32_t)(n * 128 + cu * 16), Vt, go);
    }
  };

  // stage Q tile into sK[0]; hold Q A-fragments in registers (6 k-steps of 32)
  issueK(0, q0);
  s_wait_async<0>();
  __syncthreads();
  AF qf[6];
  {
    const uint32_t* pQ = reinterpret_cast<const uint32_t*>(&sK[0][0] + (wid * 16 + row16) * DQK_);
    #pragma unroll
    for (int ks = 0; ks < 6; ++ks) {
      #pragma unroll
      for (int p = 0; p < 8; ++p)
        qf[ks].u[p] = pQ[ks * 16 + p + (p >= 4 ? 4 : 0) + hi * 4];
    }
  }
  __syncthreads();

  float mrow[8], lrow[8];
  v8f oacc[8] = {};
  #pragma unroll
  for (int r = 0; r < 8; ++r) { mrow[r] = -1e30f; lrow[r] = 0.0f; }

  const int nkb = blockIdx.x + 1;   // causal: keys up to q0+63
  issueK(0, 0);
  issueV(0, 0);
  for (int kb = 0; kb < nkb; ++kb) {
    const int cur = kb & 1;
    const int k0 = kb * 64;
    if (kb + 1 < nkb) {
      issueK(cur ^ 1, k0 + 64);
      issueV(cur ^ 1, k0 + 64);
      s_wait_async<20>();
    } else {
      s_wait_async<0>();
    }
    __syncthreads();

    // scores: S(16x64) = Q(16x192) @ K^T(192x64), 24 WMMAs
    v8f sacc[4] = {};
    #pragma unroll
    for (int ks = 0; ks < 6; ++ks) {
      #pragma unroll
      for (int t = 0; t < 4; ++t) {
        AF bf;   // B: N=key col, K=feature
        const uint32_t* pK = reinterpret_cast<const uint32_t*>(&sK[cur][0] + (t * 16 + row16) * DQK_);
        #pragma unroll
        for (int p = 0; p < 8; ++p) bf.u[p] = pK[ks * 16 + p + (p >= 4 ? 4 : 0) + hi * 4];
        sacc[t] = wmma16(qf[ks].v, bf.v, sacc[t]);
      }
    }

    // online softmax per row (row = wid*16 + hi*8 + r)
    #pragma unroll
    for (int r = 0; r < 8; ++r) {
      const int qrow = q0 + wid * 16 + hi * 8 + r;
      float sv[4];
      float rm = -1e30f;
      #pragma unroll
      for (int t = 0; t < 4; ++t) {
        float s = sacc[t][r] * SCALE_F;
        const int key = k0 + t * 16 + row16;
        if (key > qrow) s = -1e30f;   // only bites on diagonal block
        sv[t] = s;
        rm = fmaxf(rm, s);
      }
      #pragma unroll
      for (int xm = 1; xm <= 8; xm <<= 1) rm = fmaxf(rm, __shfl_xor(rm, xm, 32));
      const float mn    = fmaxf(mrow[r], rm);
      const float alpha = __expf(mrow[r] - mn);
      float rs = 0.0f;
      #pragma unroll
      for (int t = 0; t < 4; ++t) {
        const float p = __expf(sv[t] - mn);
        rs += p;
        sP[(wid * 16 + hi * 8 + r) * 64 + t * 16 + row16] = (_Float16)p;
      }
      #pragma unroll
      for (int xm = 1; xm <= 8; xm <<= 1) rs += __shfl_xor(rs, xm, 32);
      lrow[r] = lrow[r] * alpha + rs;
      mrow[r] = mn;
      #pragma unroll
      for (int t = 0; t < 8; ++t) oacc[t][r] *= alpha;
    }

    // O += P(16x64) @ V(64x128), 16 WMMAs (per-wave sP: in-order LDS within wave)
    #pragma unroll
    for (int ks = 0; ks < 2; ++ks) {
      AF pf;
      const uint32_t* pP = reinterpret_cast<const uint32_t*>(sP + (wid * 16 + row16) * 64 + ks * 32);
      #pragma unroll
      for (int p = 0; p < 8; ++p) pf.u[p] = pP[p + (p >= 4 ? 4 : 0) + hi * 4];
      #pragma unroll
      for (int t = 0; t < 8; ++t) {
        AF vf;   // B: N = v col, K = key within block
        const uint32_t* pV = reinterpret_cast<const uint32_t*>(&sV[cur][0] + (t * 16 + row16) * 64 + ks * 32);
        #pragma unroll
        for (int p = 0; p < 8; ++p) vf.u[p] = pV[p + (p >= 4 ? 4 : 0) + hi * 4];
        oacc[t] = wmma16(pf.v, vf.v, oacc[t]);
      }
    }
    __syncthreads();
  }

  // normalize + store to attn_out[b, s, h*128 + n]
  #pragma unroll
  for (int r = 0; r < 8; ++r) {
    const int s = q0 + wid * 16 + hi * 8 + r;
    const float inv = 1.0f / lrow[r];
    _Float16* orow = O + ((size_t)(b * S_ + s)) * (NH_ * DHV_) + h * DHV_;
    #pragma unroll
    for (int t = 0; t < 8; ++t)
      orow[t * 16 + row16] = (_Float16)(oacc[t][r] * inv);
  }
}

// ---------------------------------------------------------------- launch
extern "C" void kernel_launch(void* const* d_in, const int* in_sizes, int n_in,
                              void* d_out, int out_size, void* d_ws, size_t ws_size,
                              hipStream_t stream)
{
  (void)in_sizes; (void)n_in; (void)out_size; (void)ws_size;
  const float* x     = (const float*)d_in[0];
  const float* freq  = (const float*)d_in[1];
  // d_in[2] = mask (implicit causal)
  const float* w_dq  = (const float*)d_in[3];
  const float* b_dq  = (const float*)d_in[4];
  const float* w_uq  = (const float*)d_in[5];
  const float* b_uq  = (const float*)d_in[6];
  const float* w_qr  = (const float*)d_in[7];
  const float* b_qr  = (const float*)d_in[8];
  const float* w_dkv = (const float*)d_in[9];
  const float* b_dkv = (const float*)d_in[10];
  const float* w_ukv = (const float*)d_in[11];
  const float* b_ukv = (const float*)d_in[12];
  const float* w_kr  = (const float*)d_in[13];
  const float* b_kr  = (const float*)d_in[14];
  const float* w_o   = (const float*)d_in[15];
  const float* b_o   = (const float*)d_in[16];

  const size_t M = (size_t)B_ * S_;   // 4096
  _Float16* ws = (_Float16*)d_ws;
  size_t off = 0;
  auto take = [&](size_t n) { _Float16* p = ws + off; off += (n + 7) & ~(size_t)7; return p; };

  _Float16* x_h    = take(M * H_);
  _Float16* wdq_t  = take((size_t)H_ * DCQ_);
  _Float16* wuq_t  = take((size_t)DCQ_ * NH_ * DH_);
  _Float16* wqr_t  = take((size_t)DCQ_ * NH_ * DR_);
  _Float16* wdkv_t = take((size_t)H_ * DCKV_);
  _Float16* wukv_t = take((size_t)DCKV_ * NH_ * (DH_ + DHV_));
  _Float16* wkr_t  = take((size_t)H_ * DR_);
  _Float16* wo_t   = take((size_t)NH_ * DHV_ * H_);
  _Float16* cq_h   = take(M * DCQ_);
  _Float16* qc_h   = take(M * NH_ * DH_);
  _Float16* qr_h   = take(M * NH_ * DR_);
  _Float16* ckv_h  = take(M * DCKV_);
  _Float16* kvc_h  = take(M * NH_ * (DH_ + DHV_));
  _Float16* krr_h  = take(M * DR_);
  _Float16* q_h    = take((size_t)B_ * NH_ * S_ * DQK_);
  _Float16* k_h    = take((size_t)B_ * NH_ * S_ * DQK_);
  _Float16* vt_h   = take((size_t)B_ * NH_ * DHV_ * S_);
  _Float16* ao_h   = take(M * NH_ * DHV_);

  cvt_f32_f16<<<dim3((unsigned)((M * H_ + 255) / 256)), dim3(256), 0, stream>>>(x, x_h, M * H_);
  auto cvtT = [&](const float* s, _Float16* d, int K, int N) {
    size_t n = (size_t)K * N;
    cvt_f32_f16_t<<<dim3((unsigned)((n + 255) / 256)), dim3(256), 0, stream>>>(s, d, K, N);
  };
  cvtT(w_dq,  wdq_t,  H_,   DCQ_);
  cvtT(w_uq,  wuq_t,  DCQ_, NH_ * DH_);
  cvtT(w_qr,  wqr_t,  DCQ_, NH_ * DR_);
  cvtT(w_dkv, wdkv_t, H_,   DCKV_);
  cvtT(w_ukv, wukv_t, DCKV_, NH_ * (DH_ + DHV_));
  cvtT(w_kr,  wkr_t,  H_,   DR_);
  cvtT(w_o,   wo_t,   NH_ * DHV_, H_);

  auto gemm16 = [&](const _Float16* A, const _Float16* Bt, const float* bias,
                    _Float16* C, int N, int K) {
    gemm64<false><<<dim3(N / 64, (int)M / 64), dim3(256), 0, stream>>>(A, Bt, bias, (void*)C, (int)M, N, K);
  };
  gemm16(x_h,   wdq_t,  b_dq,  cq_h,  DCQ_,               H_);     // c_q
  gemm16(cq_h,  wuq_t,  b_uq,  qc_h,  NH_ * DH_,          DCQ_);   // q_c
  gemm16(cq_h,  wqr_t,  b_qr,  qr_h,  NH_ * DR_,          DCQ_);   // q_r (pre-rope)
  gemm16(x_h,   wdkv_t, b_dkv, ckv_h, DCKV_,              H_);     // c_kv
  gemm16(ckv_h, wukv_t, b_ukv, kvc_h, NH_ * (DH_ + DHV_), DCKV_);  // kv_c
  gemm16(x_h,   wkr_t,  b_kr,  krr_h, DR_,                H_);     // k_r (pre-rope)

  {
    const int tq = B_ * NH_ * S_ * DQK_;
    build_q<<<dim3((tq + 255) / 256), dim3(256), 0, stream>>>(qc_h, qr_h, freq, q_h);
    build_k<<<dim3((tq + 255) / 256), dim3(256), 0, stream>>>(kvc_h, krr_h, freq, k_h);
    const int tv = B_ * NH_ * DHV_ * S_;
    build_vt<<<dim3((tv + 255) / 256), dim3(256), 0, stream>>>(kvc_h, vt_h);
  }

  flash_attn<<<dim3(S_ / 64, NH_, B_), dim3(128), 0, stream>>>(q_h, k_h, vt_h, ao_h);

  gemm64<true><<<dim3(H_ / 64, (int)M / 64), dim3(256), 0, stream>>>(
      ao_h, wo_t, b_o, d_out, (int)M, H_, NH_ * DHV_);
}
